// ETLSTMTrain_35021163331767
// MI455X (gfx1250) — compile-verified, matrix operand order checked by reading, not run on hardware
//
#include <hip/hip_runtime.h>
#include <hip/hip_bf16.h>

// Problem constants (fixed by the reference harness)
#define KN0      60000
#define KN1      10000
#define KE       100000
#define KT       8
#define KNODE_IN 128
#define KH       128
#define KEDGE_IN 16
#define KTH      16
#define KF       32       // EDGE_IN + TH
#define KG4H     512      // 4*H

typedef __attribute__((ext_vector_type(16))) __bf16 v16bf;
typedef __attribute__((ext_vector_type(8)))  float  v8f;

union AFrag { v16bf v; unsigned int u[8]; };

__device__ __forceinline__ v8f zero8() {
  v8f z;
#pragma unroll
  for (int i = 0; i < 8; ++i) z[i] = 0.f;
  return z;
}

__device__ __forceinline__ unsigned short f2bf(float f) {
  unsigned int u = __float_as_uint(f);
  u += 0x7fffu + ((u >> 16) & 1u);        // round-to-nearest-even
  return (unsigned short)(u >> 16);
}

__device__ __forceinline__ float sigm(float x) { return 1.f / (1.f + __expf(-x)); }

// A-fragment (16xK tile, bf16, M-major buffer with row stride `stride`)
// ISA 16-bit A 16x32 layout: lane<16 -> M=lane, v0..3 = K{0..7}, v4..7 = K{16..23};
// lane>=16 -> same M, K offset +8.
__device__ __forceinline__ AFrag load_afrag(const unsigned short* buf, int stride,
                                            int kt, int lane) {
  AFrag a;
  const int hi = (lane >> 4) & 1;
  const int m  = lane & 15;
  const int kb = kt * 32 + (hi ? 8 : 0);
#pragma unroll
  for (int r = 0; r < 8; ++r) {
    const int kk = kb + (r < 4 ? 2 * r : 16 + 2 * (r - 4));
    a.u[r] = *(const unsigned int*)(buf + m * stride + kk);
  }
  return a;
}

// B-fragment for B = W^T (W row-major [Nrows][kdim], bf16). lane maps to N column,
// K packing mirrors the A layout.
__device__ __forceinline__ AFrag load_bfrag(const unsigned short* __restrict__ W,
                                            int kdim, int n0, int kt, int lane) {
  AFrag b;
  const int hi = (lane >> 4) & 1;
  const int n  = n0 + (lane & 15);
  const int kb = kt * 32 + (hi ? 8 : 0);
#pragma unroll
  for (int r = 0; r < 8; ++r) {
    const int kk = kb + (r < 4 ? 2 * r : 16 + 2 * (r - 4));
    b.u[r] = *(const unsigned int*)(W + (size_t)n * kdim + kk);
  }
  return b;
}

__device__ __forceinline__ v8f wmma_bf16(const AFrag& a, const AFrag& b, v8f c) {
  return __builtin_amdgcn_wmma_f32_16x16x32_bf16(false, a.v, false, b.v,
                                                 (short)0, c, false, false);
}

// ---------------------------------------------------------------------------
// Kernel 0: weight prep — convert GEMM weights to bf16, fuse bih+bhh.
// ---------------------------------------------------------------------------
__global__ void prep_kernel(const float* __restrict__ Wih, const float* __restrict__ Whh,
                            const float* __restrict__ eoW, const float* __restrict__ nlW,
                            const float* __restrict__ bih, const float* __restrict__ bhh,
                            unsigned short* __restrict__ Wihb, unsigned short* __restrict__ Whhb,
                            unsigned short* __restrict__ eoWb, unsigned short* __restrict__ nlWb,
                            float* __restrict__ bsum) {
  int i = blockIdx.x * 256 + threadIdx.x;
  if (i < KG4H * KF)        { Wihb[i] = f2bf(Wih[i]); return; }
  i -= KG4H * KF;
  if (i < KG4H * KH)        { Whhb[i] = f2bf(Whh[i]); return; }
  i -= KG4H * KH;
  if (i < KH * 2 * KNODE_IN){ eoWb[i] = f2bf(eoW[i]); return; }
  i -= KH * 2 * KNODE_IN;
  if (i < KH * 2 * KNODE_IN){ nlWb[i] = f2bf(nlW[i]); return; }
  i -= KH * 2 * KNODE_IN;
  if (i < KG4H)             { bsum[i] = bih[i] + bhh[i]; }
}

// ---------------------------------------------------------------------------
// Kernel 1: per 16-edge tile: Time2Vec + LSTM(T=8, pack-padded last) + message
// MLP + atomic scatter into hd[N1][128].  128 threads = 4 waves; wave w owns
// hidden units [32w, 32w+32).
// ---------------------------------------------------------------------------
__global__ __launch_bounds__(128) void edge_lstm_msg_kernel(
    const float* __restrict__ nf, const float* __restrict__ e,
    const float* __restrict__ e_times,
    const float* __restrict__ tw0, const float* __restrict__ tb0,
    const float* __restrict__ tw,  const float* __restrict__ tb,
    const float* __restrict__ eob, const float* __restrict__ bsum,
    const unsigned short* __restrict__ Wihb,
    const unsigned short* __restrict__ Whhb,
    const unsigned short* __restrict__ eoWb,
    const int* __restrict__ e_len, const int* __restrict__ src,
    const int* __restrict__ dst, float* __restrict__ hd) {
  __shared__ __align__(16) unsigned short x_sm[KT][16][KF];     // 8 KB
  __shared__ __align__(16) unsigned short nf_sm[16][KNODE_IN];  // 4 KB
  __shared__ __align__(16) unsigned short h_sm[16][KH];         // 4 KB
  __shared__ int elen_sm[16];
  __shared__ int dst_sm[16];

  const int tid = threadIdx.x;
  const int e0  = blockIdx.x * 16;

  { // Stage x = [e | Time2Vec(t)] as bf16; one (edge,t) pair per thread.
    const int el = tid >> 3, t = tid & 7;
    const int ge = e0 + el;
    const float tt = e_times[(size_t)ge * KT + t];
    const float* ep = e + ((size_t)ge * KT + t) * KEDGE_IN;
#pragma unroll
    for (int k = 0; k < KEDGE_IN; ++k) x_sm[t][el][k] = f2bf(ep[k]);
#pragma unroll
    for (int j = 0; j < KTH - 1; ++j)
      x_sm[t][el][KEDGE_IN + j] = f2bf(__sinf(tt * tw[j] + tb[j]));
    x_sm[t][el][KEDGE_IN + KTH - 1] = f2bf(tt * tw0[0] + tb0[0]);
  }
  for (int i = tid; i < 16 * KNODE_IN; i += 128) {  // gather src node features
    const int el = i >> 7, k = i & 127;
    nf_sm[el][k] = f2bf(nf[(size_t)src[e0 + el] * KNODE_IN + k]);
    ((unsigned short*)h_sm)[i] = 0;                 // h0 = 0
  }
  if (tid < 16) { elen_sm[tid] = e_len[e0 + tid]; dst_sm[tid] = dst[e0 + tid]; }
  __syncthreads();

  const int w = tid >> 5, lane = tid & 31;
  const int hi = (lane >> 4) & 1, l15 = lane & 15;

  int elen_r[8], dst_r[8];
#pragma unroll
  for (int r = 0; r < 8; ++r) {
    const int M = r + hi * 8;
    elen_r[r] = elen_sm[M];
    dst_r[r]  = dst_sm[M];
  }

  v8f c_st[2], hlast[2];
  c_st[0] = zero8(); c_st[1] = zero8();
  hlast[0] = zero8(); hlast[1] = zero8();

  for (int t = 0; t < KT; ++t) {
    __syncthreads();                       // h_{t-1} writes visible
    AFrag aH[4];
#pragma unroll
    for (int kt = 0; kt < 4; ++kt) aH[kt] = load_afrag(&h_sm[0][0], KH, kt, lane);
    AFrag aX = load_afrag(&x_sm[t][0][0], KF, 0, lane);
    __syncthreads();                       // everyone done reading h_sm

    v8f gacc[8];                           // gate g = q>>1, subtile s = q&1
#pragma unroll
    for (int q = 0; q < 8; ++q) {
      const int g = q >> 1, s = q & 1;
      const int n0 = g * KH + w * 32 + s * 16;
      v8f acc = zero8();
      AFrag bX = load_bfrag(Wihb, KF, n0, 0, lane);
      acc = wmma_bf16(aX, bX, acc);        // x_t @ Wih^T slice
#pragma unroll
      for (int kt = 0; kt < 4; ++kt) {
        AFrag bH = load_bfrag(Whhb, KH, n0, kt, lane);
        acc = wmma_bf16(aH[kt], bH, acc);  // h_{t-1} @ Whh^T slice
      }
      const float bb = bsum[n0 + l15];
#pragma unroll
      for (int r = 0; r < 8; ++r) acc[r] += bb;
      gacc[q] = acc;
    }
#pragma unroll
    for (int s = 0; s < 2; ++s) {
#pragma unroll
      for (int r = 0; r < 8; ++r) {
        const float iv = sigm(gacc[0 + s][r]);
        const float fv = sigm(gacc[2 + s][r]);
        const float gv = tanhf(gacc[4 + s][r]);
        const float ov = sigm(gacc[6 + s][r]);
        const float cc = fv * c_st[s][r] + iv * gv;
        c_st[s][r] = cc;
        const float hv = ov * tanhf(cc);
        h_sm[r + hi * 8][w * 32 + s * 16 + l15] = f2bf(hv);
        if (t < elen_r[r]) hlast[s][r] = hv;   // pack_padded: keep h at e_len-1
      }
    }
  }

  // Park h_last in h_sm (each wave rewrites its own columns), then message GEMM:
  // m = relu([nf_src | h_last] @ eoW^T + eob), scattered with f32 atomics.
#pragma unroll
  for (int s = 0; s < 2; ++s)
#pragma unroll
    for (int r = 0; r < 8; ++r)
      h_sm[r + hi * 8][w * 32 + s * 16 + l15] = f2bf(hlast[s][r]);
  __syncthreads();

  AFrag aM[8];
#pragma unroll
  for (int kt = 0; kt < 4; ++kt) {
    aM[kt]     = load_afrag(&nf_sm[0][0], KNODE_IN, kt, lane);
    aM[4 + kt] = load_afrag(&h_sm[0][0],  KH,       kt, lane);
  }
#pragma unroll
  for (int s = 0; s < 2; ++s) {
    const int n0 = w * 32 + s * 16;
    v8f acc = zero8();
#pragma unroll
    for (int kt = 0; kt < 8; ++kt) {
      AFrag b = load_bfrag(eoWb, 2 * KNODE_IN, n0, kt, lane);
      acc = wmma_bf16(aM[kt], b, acc);
    }
    const float bb = eob[n0 + l15];
#pragma unroll
    for (int r = 0; r < 8; ++r) {
      float m = acc[r] + bb;
      m = m > 0.f ? m : 0.f;
      atomicAdd(&hd[(size_t)dst_r[r] * KH + n0 + l15], m);
    }
  }
}

// ---------------------------------------------------------------------------
// Kernel 2: per 16-node tile: self_h_tmp GEMM, (hd - tmp)*norm, node-update
// GEMM + relu, fc head.  Same 4-wave WMMA tiling as the edge kernel.
// ---------------------------------------------------------------------------
__global__ __launch_bounds__(128) void node_update_kernel(
    const float* __restrict__ nf, const float* __restrict__ subg,
    const float* __restrict__ eob, const float* __restrict__ nlb,
    const float* __restrict__ fcW, const float* __restrict__ fcb,
    const int* __restrict__ layer_nid, const float* __restrict__ hd,
    const unsigned short* __restrict__ eoWb,
    const unsigned short* __restrict__ nlWb, float* __restrict__ out) {
  __shared__ __align__(16) unsigned short sh_sm[16][KNODE_IN];  // self_h bf16
  __shared__ __align__(16) unsigned short hc_sm[16][KH];        // normalized h bf16
  __shared__ __align__(16) float z_sm[16][KH];
  __shared__ float sn_sm[16];

  const int tid = threadIdx.x;
  const int nb  = blockIdx.x * 16;

  for (int i = tid; i < 16 * KNODE_IN; i += 128) {
    const int nl = i >> 7, k = i & 127;
    sh_sm[nl][k] = f2bf(nf[(size_t)layer_nid[nb + nl] * KNODE_IN + k]);
  }
  if (tid < 16) sn_sm[tid] = subg[nb + tid];
  __syncthreads();

  const int w = tid >> 5, lane = tid & 31;
  const int hi = (lane >> 4) & 1, l15 = lane & 15;

  AFrag aS[4];
#pragma unroll
  for (int kt = 0; kt < 4; ++kt) aS[kt] = load_afrag(&sh_sm[0][0], KNODE_IN, kt, lane);

  // self_h_tmp = [self_h | 0] @ eoW^T + eob  (only first 128 cols of eoW matter)
#pragma unroll
  for (int s = 0; s < 2; ++s) {
    const int n0 = w * 32 + s * 16;
    v8f acc = zero8();
#pragma unroll
    for (int kt = 0; kt < 4; ++kt) {
      AFrag b = load_bfrag(eoWb, 2 * KNODE_IN, n0, kt, lane);
      acc = wmma_bf16(aS[kt], b, acc);
    }
    const float bb = eob[n0 + l15];
#pragma unroll
    for (int r = 0; r < 8; ++r) {
      const int M = r + hi * 8;
      const float tmp = acc[r] + bb;
      const float hv = (hd[(size_t)(nb + M) * KH + n0 + l15] - tmp) * sn_sm[M];
      hc_sm[M][n0 + l15] = f2bf(hv);
    }
  }
  __syncthreads();

  AFrag aC[4];
#pragma unroll
  for (int kt = 0; kt < 4; ++kt) aC[kt] = load_afrag(&hc_sm[0][0], KH, kt, lane);

  // z = relu([self_h | h] @ nlW^T + nlb)
#pragma unroll
  for (int s = 0; s < 2; ++s) {
    const int n0 = w * 32 + s * 16;
    v8f acc = zero8();
#pragma unroll
    for (int kt = 0; kt < 8; ++kt) {
      AFrag a = (kt < 4) ? aS[kt] : aC[kt - 4];
      AFrag b = load_bfrag(nlWb, 2 * KNODE_IN, n0, kt, lane);
      acc = wmma_bf16(a, b, acc);
    }
    const float bb = nlb[n0 + l15];
#pragma unroll
    for (int r = 0; r < 8; ++r) {
      float z = acc[r] + bb;
      z_sm[r + hi * 8][n0 + l15] = z > 0.f ? z : 0.f;
    }
  }
  __syncthreads();

  // fc: 2 outputs per node, 32 dots of length 128.
  if (tid < 32) {
    const int nl = tid >> 1, c = tid & 1;
    float s = fcb[c];
#pragma unroll 8
    for (int k = 0; k < KH; ++k) s += fcW[c * KH + k] * z_sm[nl][k];
    out[(size_t)(nb + nl) * 2 + c] = s;
  }
}

// ---------------------------------------------------------------------------
extern "C" void kernel_launch(void* const* d_in, const int* in_sizes, int n_in,
                              void* d_out, int out_size, void* d_ws, size_t ws_size,
                              hipStream_t stream) {
  const float* nf      = (const float*)d_in[0];
  const float* e       = (const float*)d_in[1];
  const float* e_times = (const float*)d_in[2];
  const float* subg    = (const float*)d_in[3];
  const float* tw0     = (const float*)d_in[4];
  const float* tb0     = (const float*)d_in[5];
  const float* tw      = (const float*)d_in[6];
  const float* tb      = (const float*)d_in[7];
  const float* Wih     = (const float*)d_in[8];
  const float* Whh     = (const float*)d_in[9];
  const float* bih     = (const float*)d_in[10];
  const float* bhh     = (const float*)d_in[11];
  const float* eoW     = (const float*)d_in[12];
  const float* eob     = (const float*)d_in[13];
  const float* nlW     = (const float*)d_in[14];
  const float* nlb     = (const float*)d_in[15];
  const float* fcW     = (const float*)d_in[16];
  const float* fcb     = (const float*)d_in[17];
  const int* e_len     = (const int*)d_in[18];
  const int* src       = (const int*)d_in[19];
  const int* dst       = (const int*)d_in[20];
  const int* layer_nid = (const int*)d_in[21];
  float* out = (float*)d_out;

  const int Eedges = in_sizes[18];   // 100000
  const int Nnodes = in_sizes[3];    // 10000

  // Workspace layout (all offsets 256B-aligned)
  char* ws = (char*)d_ws;
  size_t off = 0;
  float* hd = (float*)(ws + off);                 off += ((size_t)Nnodes * KH * 4 + 255) & ~(size_t)255;
  unsigned short* Wihb = (unsigned short*)(ws + off); off += ((size_t)KG4H * KF * 2 + 255) & ~(size_t)255;
  unsigned short* Whhb = (unsigned short*)(ws + off); off += ((size_t)KG4H * KH * 2 + 255) & ~(size_t)255;
  unsigned short* eoWb = (unsigned short*)(ws + off); off += ((size_t)KH * 2 * KNODE_IN * 2 + 255) & ~(size_t)255;
  unsigned short* nlWb = (unsigned short*)(ws + off); off += ((size_t)KH * 2 * KNODE_IN * 2 + 255) & ~(size_t)255;
  float* bsum = (float*)(ws + off);               off += ((size_t)KG4H * 4 + 255) & ~(size_t)255;

  hipMemsetAsync(hd, 0, (size_t)Nnodes * KH * sizeof(float), stream);

  const int prep_elems = KG4H * KF + KG4H * KH + 2 * (KH * 2 * KNODE_IN) + KG4H;
  prep_kernel<<<(prep_elems + 255) / 256, 256, 0, stream>>>(
      Wih, Whh, eoW, nlW, bih, bhh, Wihb, Whhb, eoWb, nlWb, bsum);

  edge_lstm_msg_kernel<<<Eedges / 16, 128, 0, stream>>>(
      nf, e, e_times, tw0, tb0, tw, tb, eob, bsum,
      Wihb, Whhb, eoWb, e_len, src, dst, hd);

  node_update_kernel<<<Nnodes / 16, 128, 0, stream>>>(
      nf, subg, eob, nlb, fcW, fcb, layer_nid, hd, eoWb, nlWb, out);
}
